// SimpleAttention_82635170775303
// MI455X (gfx1250) — matrix operand order repaired and checked by reading, not compiled, not run on hardware
//
#include <hip/hip_runtime.h>

// ---------------------------------------------------------------- types ----
typedef __attribute__((ext_vector_type(16))) __bf16 v16bf;
typedef __attribute__((ext_vector_type(8)))  __bf16 v8bf;
typedef __attribute__((ext_vector_type(4)))  __bf16 v4bf;
typedef __attribute__((ext_vector_type(4)))  float  v4f;
typedef __attribute__((ext_vector_type(8)))  float  v8f;

#define B_   2
#define T_   2048
#define E_   1024
#define H_   16
#define D_   64
#define F3_  3072
#define BT_  4096   // B_*T_

__device__ __forceinline__ v16bf cat8(v8bf lo, v8bf hi) {
  return __builtin_shufflevector(lo, hi, 0,1,2,3,4,5,6,7,8,9,10,11,12,13,14,15);
}

// A-operand fragment (16xK, rows of `base`): lanes 0-15 row=lane, K chunks
// {k0+0..7, k0+16..23}; lanes 16-31 row=lane-16, K chunks {k0+8..15, k0+24..31}.
__device__ __forceinline__ v16bf load_a_frag(const __bf16* base, int stride,
                                             int row0, int k0, int lane) {
  int half = lane >> 4;
  int r    = lane & 15;
  const __bf16* p = base + (size_t)(row0 + r) * stride + k0 + half * 8;
  v8bf lo = *(const v8bf*)p;
  v8bf hi = *(const v8bf*)(p + 16);
  return cat8(lo, hi);
}

// B-operand fragment (Kx16, "cols" are rows of `base`): lanes 0-15 col=lane,
// K chunks {k0+0..7, k0+16..23}; lanes 16-31 col=lane-16, K chunks {k0+8..15, k0+24..31}.
__device__ __forceinline__ v16bf load_b_frag(const __bf16* base, int stride,
                                             int col0, int k0, int lane) {
  int half = lane >> 4;
  int c    = lane & 15;
  const __bf16* p = base + (size_t)(col0 + c) * stride + k0 + half * 8;
  v8bf lo = *(const v8bf*)p;
  v8bf hi = *(const v8bf*)(p + 16);
  return cat8(lo, hi);
}

__device__ __forceinline__ v8f wmma_bf16(v16bf a, v16bf b, v8f c) {
  return __builtin_amdgcn_wmma_f32_16x16x32_bf16(false, a, false, b,
                                                 (short)0, c, false, false);
}

// ---------------------------------------------------- fp32 -> bf16 convert ----
__global__ __launch_bounds__(256) void cvt_bf16x4(const float* __restrict__ src,
                                                  __bf16* __restrict__ dst, int n4) {
  int i = blockIdx.x * blockDim.x + threadIdx.x;
  if (i < n4) {
    v4f v = *(const v4f*)(src + i * 4);
    v4bf o;
    o.x = (__bf16)v.x; o.y = (__bf16)v.y; o.z = (__bf16)v.z; o.w = (__bf16)v.w;
    *(v4bf*)(dst + i * 4) = o;
  }
}

// --------------------------------------------------------------- QKV GEMM ----
// QKV[bt, f] = sum_e X[bt, e] * W[f, e];  f = chunk*1024 + h*64 + d.
// One wave: 32 (rows) x 64 (f) tile -> 8 WMMAs per 12 b128 loads per k-step.
// Q scaled by 1/sqrt(D).  V stored transposed [b,h,d,t].
__global__ __launch_bounds__(128) void gemm_qkv(const __bf16* __restrict__ X,
                                                const __bf16* __restrict__ W,
                                                __bf16* __restrict__ Qh,
                                                __bf16* __restrict__ Kh,
                                                __bf16* __restrict__ Vt) {
  int wave = blockIdx.x * 4 + (threadIdx.x >> 5);
  int lane = threadIdx.x & 31;
  int tm = wave & 127;       // 128 row tiles of 32
  int tn = wave >> 7;        // 48 f tiles of 64
  int row0 = tm * 32;
  int f0   = tn * 64;

  v8f acc0[4] = {}, acc1[4] = {};
  for (int k = 0; k < E_; k += 32) {
    v16bf a0 = load_a_frag(X, E_, row0,      k, lane);
    v16bf a1 = load_a_frag(X, E_, row0 + 16, k, lane);
    if (k + 64 < E_)   // stream-ahead hint for X (weights stay L2-hot)
      __builtin_prefetch(X + (size_t)(row0 + (lane & 15)) * E_ + k + 64, 0, 3);
#pragma unroll
    for (int n = 0; n < 4; ++n) {
      v16bf b = load_b_frag(W, E_, f0 + n * 16, k, lane);
      acc0[n] = wmma_bf16(a0, b, acc0[n]);
      acc1[n] = wmma_bf16(a1, b, acc1[n]);
    }
  }

  int chunk = f0 >> 10;            // 0:Q 1:K 2:V
  int h     = (f0 & 1023) >> 6;    // head
  int half = lane >> 4, col = lane & 15;
  float scale = (chunk == 0) ? 0.125f : 1.0f;
#pragma unroll
  for (int blk = 0; blk < 2; ++blk) {
#pragma unroll
    for (int n = 0; n < 4; ++n) {
#pragma unroll
      for (int r = 0; r < 8; ++r) {
        float av = (blk == 0) ? acc0[n][r] : acc1[n][r];
        int row = row0 + blk * 16 + r + half * 8;   // global bt row
        int b   = row >> 11;
        int t   = row & 2047;
        int d   = n * 16 + col;
        __bf16 v = (__bf16)(av * scale);
        size_t bh = (size_t)(b * H_ + h);
        if (chunk == 0)      Qh[(bh * T_ + t) * D_ + d] = v;
        else if (chunk == 1) Kh[(bh * T_ + t) * D_ + d] = v;
        else                 Vt[(bh * D_ + d) * T_ + t] = v;
      }
    }
  }
}

// ---------------------------------------------------------- flash attention ----
// One wave per (b, h, 16-query tile); 64 keys per iteration (4 S subtiles)
// to amortize the cross-lane softmax reductions over more keys.
__global__ __launch_bounds__(128) void attn_flash(const __bf16* __restrict__ Qh,
                                                  const __bf16* __restrict__ Kh,
                                                  const __bf16* __restrict__ Vt,
                                                  __bf16* __restrict__ Xa) {
  __shared__ __bf16 plds[4][16][64];           // per-wave P staging (2KB each)
  int wv   = blockIdx.x * 4 + (threadIdx.x >> 5);
  int lane = threadIdx.x & 31;
  int wslot = threadIdx.x >> 5;
  int qt = wv & 127;                 // 128 query tiles
  int bh = wv >> 7;                  // 32 (b,h) pairs
  int q0 = qt * 16;
  const __bf16* Qb = Qh + (size_t)bh * T_ * D_;
  const __bf16* Kb = Kh + (size_t)bh * T_ * D_;
  const __bf16* Vb = Vt + (size_t)bh * D_ * T_;
  int half = lane >> 4, col = lane & 15;

  v16bf qa0 = load_a_frag(Qb, D_, q0, 0, lane);
  v16bf qa1 = load_a_frag(Qb, D_, q0, 32, lane);

  float m[8], l[8];
  v8f o[4] = {};
#pragma unroll
  for (int r = 0; r < 8; ++r) { m[r] = -1e30f; l[r] = 0.0f; }

  for (int kb = 0; kb <= q0 + 15; kb += 64) {
    // ---- S = Q K^T  (16 queries x 64 keys, 8 WMMAs) ----
    v8f s[4];
#pragma unroll
    for (int j = 0; j < 4; ++j) {
      v8f sj = {};
      v16bf blo = load_b_frag(Kb, D_, kb + j * 16, 0, lane);
      v16bf bhi = load_b_frag(Kb, D_, kb + j * 16, 32, lane);
      sj = wmma_bf16(qa0, blo, sj);
      sj = wmma_bf16(qa1, bhi, sj);
      s[j] = sj;
    }
    // ---- causal mask + online softmax (row r lives in one 16-lane half) ----
#pragma unroll
    for (int r = 0; r < 8; ++r) {
      int qpos = q0 + r + half * 8;
      float a[4];
#pragma unroll
      for (int j = 0; j < 4; ++j) {
        a[j] = s[j][r];
        if (kb + j * 16 + col > qpos) a[j] = -1e30f;
      }
      float vm = fmaxf(fmaxf(a[0], a[1]), fmaxf(a[2], a[3]));
#pragma unroll
      for (int off = 8; off >= 1; off >>= 1)
        vm = fmaxf(vm, __shfl_xor(vm, off, 32));
      float mn = fmaxf(m[r], vm);
      float sc = __expf(m[r] - mn);
      float p[4], rs = 0.0f;
#pragma unroll
      for (int j = 0; j < 4; ++j) { p[j] = __expf(a[j] - mn); rs += p[j]; }
#pragma unroll
      for (int off = 8; off >= 1; off >>= 1)
        rs += __shfl_xor(rs, off, 32);
      l[r] = l[r] * sc + rs;
      m[r] = mn;
#pragma unroll
      for (int n = 0; n < 4; ++n) o[n][r] *= sc;
#pragma unroll
      for (int j = 0; j < 4; ++j)                       // C-layout -> LDS
        plds[wslot][r + half * 8][j * 16 + col] = (__bf16)p[j];
    }
    asm volatile("s_wait_dscnt 0" ::: "memory");        // wave-local LDS RAW
    // ---- O += P V  (P read back in A-layout, V B-operand contiguous) ----
    v16bf pa0 = load_a_frag(&plds[wslot][0][0], 64, 0, 0, lane);
    v16bf pa1 = load_a_frag(&plds[wslot][0][0], 64, 0, 32, lane);
#pragma unroll
    for (int n = 0; n < 4; ++n) {
      v16bf vb0 = load_b_frag(Vb, T_, n * 16, kb, lane);
      v16bf vb1 = load_b_frag(Vb, T_, n * 16, kb + 32, lane);
      o[n] = wmma_bf16(pa0, vb0, o[n]);
      o[n] = wmma_bf16(pa1, vb1, o[n]);
    }
  }

  // ---- finalize: O/l -> Xa[b, t, h*64+d] ----
  int b = bh >> 4, h = bh & 15;
#pragma unroll
  for (int n = 0; n < 4; ++n) {
#pragma unroll
    for (int r = 0; r < 8; ++r) {
      int t = q0 + r + half * 8;
      float v = o[n][r] / l[r];
      Xa[((size_t)(b * T_ + t)) * E_ + h * D_ + n * 16 + col] = (__bf16)v;
    }
  }
}

// ------------------------------------------------------------ output GEMM ----
// One wave: 32 (rows) x 64 (f) tile, fp32 output.
__global__ __launch_bounds__(128) void gemm_out(const __bf16* __restrict__ X,
                                                const __bf16* __restrict__ W,
                                                float* __restrict__ out) {
  int wave = blockIdx.x * 4 + (threadIdx.x >> 5);
  int lane = threadIdx.x & 31;
  int tm = wave & 127;       // 128 row tiles of 32
  int tn = wave >> 7;        // 16 f tiles of 64
  int row0 = tm * 32;
  int f0   = tn * 64;

  v8f acc0[4] = {}, acc1[4] = {};
  for (int k = 0; k < E_; k += 32) {
    v16bf a0 = load_a_frag(X, E_, row0,      k, lane);
    v16bf a1 = load_a_frag(X, E_, row0 + 16, k, lane);
    if (k + 64 < E_)
      __builtin_prefetch(X + (size_t)(row0 + (lane & 15)) * E_ + k + 64, 0, 3);
#pragma unroll
    for (int n = 0; n < 4; ++n) {
      v16bf b = load_b_frag(W, E_, f0 + n * 16, k, lane);
      acc0[n] = wmma_bf16(a0, b, acc0[n]);
      acc1[n] = wmma_bf16(a1, b, acc1[n]);
    }
  }
  int half = lane >> 4, col = lane & 15;
#pragma unroll
  for (int blk = 0; blk < 2; ++blk) {
#pragma unroll
    for (int n = 0; n < 4; ++n) {
#pragma unroll
      for (int r = 0; r < 8; ++r) {
        float av = (blk == 0) ? acc0[n][r] : acc1[n][r];
        int row = row0 + blk * 16 + r + half * 8;
        out[(size_t)row * E_ + f0 + n * 16 + col] = av;
      }
    }
  }
}

// ---------------------------------------------------------------- launcher ----
extern "C" void kernel_launch(void* const* d_in, const int* in_sizes, int n_in,
                              void* d_out, int out_size, void* d_ws, size_t ws_size,
                              hipStream_t stream) {
  const float* q     = (const float*)d_in[0];   // [B,T,E]
  // d_in[1] (key), d_in[2] (value) are unused by the reference.
  const float* w_in  = (const float*)d_in[3];   // [3E,E]
  const float* w_out = (const float*)d_in[4];   // [E,E]
  float* out = (float*)d_out;

  char* ws = (char*)d_ws;
  __bf16* qbf  = (__bf16*)(ws + 0);           // 8 MB   [BT,E]
  __bf16* w1bf = (__bf16*)(ws + 8388608);     // 6 MB   [3E,E]
  __bf16* w2bf = (__bf16*)(ws + 14680064);    // 2 MB   [E,E]
  __bf16* Qh   = (__bf16*)(ws + 16777216);    // 8 MB   [B,H,T,D] (pre-scaled)
  __bf16* Kh   = (__bf16*)(ws + 25165824);    // 8 MB   [B,H,T,D]
  __bf16* Vt   = (__bf16*)(ws + 33554432);    // 8 MB   [B,H,D,T]
  __bf16* Xa   = (__bf16*)(ws + 41943040);    // 8 MB   [B,T,E]

  cvt_bf16x4<<<4096, 256, 0, stream>>>(q,     qbf,  BT_ * E_ / 4);
  cvt_bf16x4<<<3072, 256, 0, stream>>>(w_in,  w1bf, F3_ * E_ / 4);
  cvt_bf16x4<<<1024, 256, 0, stream>>>(w_out, w2bf, E_ * E_ / 4);

  gemm_qkv  <<<1536, 128, 0, stream>>>(qbf, w1bf, Qh, Kh, Vt);   // 6144 waves
  attn_flash<<<1024, 128, 0, stream>>>(Qh, Kh, Vt, Xa);          // 4096 waves
  gemm_out  <<< 512, 128, 0, stream>>>(Xa, w2bf, out);           // 2048 waves
}
